// ald_pic_n_18305150616068
// MI455X (gfx1250) — matrix operand order repaired and checked by reading, not compiled, not run on hardware
//
#include <hip/hip_runtime.h>
#include <hip/hip_bf16.h>

// ---------------------------------------------------------------------------
// MI455X (gfx1250, wave32) implementation.
//
// Phase 1 (rank_chain_kernel): 32 blocks x 256 threads. The reference's
//   cmul = bv@C + C@bv with C == const(-1e5) is a rank-1 update:
//   rcm[i][j] += -1e5 * (rowsum(bv)[i] + colsum(bv)[j]).
//   64 sequential steps of {top-2 over 4096, build b, row/col sums, mask}.
//
// Phase 2 (patch_mix_kernel): memory-bound (154 MB HBM traffic ~ 6.6 us floor
//   @ 23.3 TB/s). FP32 data in HBM, multiply on the f16 WMMA pipe
//   (v_wmma_f32_16x16x32_f16, f32 accumulate). One wave = one 16x64 strip.
//   Fragment assembly is vectorized:
//     - R stored TRANSPOSED as f16 in LDS with 72-half row pad (144 B, 16B-
//       aligned chunks) -> each B fragment = 2x ds_load_b128.
//     - x converted to f16 at staging -> each A fragment pair = 4x ds_load_b128.
//     - 8-wide output runs never cross the %56 boundary (56 % 8 == 0) ->
//       2x global_store_b128 per accumulator tile instead of 8 scattered b32.
// ---------------------------------------------------------------------------

typedef __attribute__((ext_vector_type(16))) _Float16 v16h;
typedef __attribute__((ext_vector_type(8)))  _Float16 v8h;
typedef __attribute__((ext_vector_type(8)))  float    v8f;
typedef __attribute__((ext_vector_type(4)))  float    v4f;

union V16U { v16h v; v8h h[2]; };

#define S64    64
#define SS     4096
#define PATCH  56
#define W448   448
#define NEGC   (-100000.0f)
#define NFRAC  0.25f
#define MTILES 196              // 3136 / 16, exact
#define BANDS  25               // ceil(196/8)
#define RPAD   72               // halves per LDS row: 144 B, keeps 16B align

// ============================ Phase 1 =====================================
__global__ __launch_bounds__(256)
void rank_chain_kernel(const float* __restrict__ params,
                       const int*  __restrict__ batch_seq,
                       const int*  __restrict__ batchsize,
                       float* __restrict__ Rws,       // [32][4096] (batch-reversed)
                       float* __restrict__ out_res)   // result section of d_out
{
    __shared__ float rcm[SS];
    __shared__ float rs[S64];
    __shared__ float cs[S64];
    __shared__ float red1[8];
    __shared__ float red2[8];

    const int tid  = threadIdx.x;
    const int lane = tid & 31;
    const int wid  = tid >> 5;
    const int rb   = blockIdx.x;                       // chain batch 0..31
    const int bs   = batchsize[0];
    const int start = batch_seq[0] * bs;

    const float* p = params + (size_t)(start + rb) * SS;

    #pragma unroll
    for (int r = 0; r < 16; ++r) rcm[r * 256 + tid] = p[r * 256 + tid];
    __syncthreads();

    // ---- softmax over axis 0 (per column j) ----
    if (tid < S64) {
        const int j = tid;
        float mx = rcm[j];
        for (int i = 1; i < S64; ++i) mx = fmaxf(mx, rcm[i * S64 + j]);
        float sum = 0.f;
        for (int i = 0; i < S64; ++i) sum += __expf(rcm[i * S64 + j] - mx);
        const float inv = 1.0f / sum;
        for (int i = 0; i < S64; ++i)
            rcm[i * S64 + j] = __expf(rcm[i * S64 + j] - mx) * inv;
    }
    __syncthreads();

    float racc[16];
    #pragma unroll
    for (int r = 0; r < 16; ++r) racc[r] = 0.f;

    for (int t = 0; t < S64; ++t) {
        float m1 = -INFINITY, m2 = -INFINITY;
        #pragma unroll
        for (int r = 0; r < 16; ++r) {
            const float v = rcm[r * 256 + tid];
            if (v > m1)      { m2 = m1; m1 = v; }
            else if (v > m2) { m2 = v; }
        }
        #pragma unroll
        for (int off = 16; off > 0; off >>= 1) {
            const float o1 = __shfl_down(m1, off, 32);
            const float o2 = __shfl_down(m2, off, 32);
            const float n1 = fmaxf(m1, o1);
            const float n2 = fmaxf(fminf(m1, o1), fmaxf(m2, o2));
            m1 = n1; m2 = n2;
        }
        if (lane == 0) { red1[wid] = m1; red2[wid] = m2; }
        if (tid < S64) { rs[tid] = 0.f; cs[tid] = 0.f; }
        __syncthreads();

        float g1 = red1[0], g2 = red2[0];
        #pragma unroll
        for (int w = 1; w < 8; ++w) {
            const float o1 = red1[w], o2 = red2[w];
            const float n1 = fmaxf(g1, o1);
            const float n2 = fmaxf(fminf(g1, o1), fmaxf(g2, o2));
            g1 = n1; g2 = n2;
        }

        const float thr    = g2 + NFRAC * (g1 - g2);
        const float invden = 1.0f / ((1.0f - NFRAC) * (g1 - g2));
        const float bmax   = (g1 - thr) * invden;      // == jnp.max(b)
        const float invbm  = 1.0f / bmax;

        #pragma unroll
        for (int r = 0; r < 16; ++r) {
            const int   e = r * 256 + tid;
            const float v = rcm[e];
            float bb = v - thr;
            bb = (bb > 0.f) ? bb * invden * invbm : 0.f;
            if (bb != 0.f) {
                atomicAdd(&rs[e >> 6], bb);
                atomicAdd(&cs[e & 63], bb);
            }
            racc[r] += bb;
        }
        __syncthreads();

        #pragma unroll
        for (int r = 0; r < 16; ++r) {
            const int e = r * 256 + tid;
            rcm[e] += NEGC * (rs[e >> 6] + cs[e & 63]);
        }
        __syncthreads();
    }

    const int bout = (bs - 1) - rb;                    // result = result_r[::-1]
    float* wsdst = Rws + (size_t)bout * SS;
    #pragma unroll
    for (int r = 0; r < 16; ++r) {
        const int   e = r * 256 + tid;
        const float v = racc[r];
        wsdst[e] = v;
        #pragma unroll
        for (int ch = 0; ch < 3; ++ch)
            out_res[((size_t)(bout * 3 + ch)) * SS + e] = v;
    }
}

// ============================ Phase 2 =====================================
__global__ __launch_bounds__(256)
void patch_mix_kernel(const float* __restrict__ x,
                      const float* __restrict__ Rws,
                      float* __restrict__ out)
{
    __shared__ _Float16 RhT[S64 * RPAD];       // R transposed: RhT[n][k]
    __shared__ _Float16 Ash[8][16 * RPAD];     // per-wave A tile: Ash[m][k]

    const int tid  = threadIdx.x;
    const int lane = tid & 31;
    const int wid  = tid >> 5;

    const int band = blockIdx.x % BANDS;
    const int bc   = blockIdx.x / BANDS;       // 0..95 == b*3 + c
    const int b    = bc / 3;

    // ---- stage R transposed as f16 (coalesced global reads, LDS scatter) ----
    const float* Rsrc = Rws + (size_t)b * SS;
    #pragma unroll
    for (int r = 0; r < 16; ++r) {
        const int e = r * 256 + tid;
        const int k = e >> 6, n = e & 63;
        RhT[n * RPAD + k] = (_Float16)Rsrc[e];
    }

    // ---- stage this wave's 16x64 A tile (f32 -> f16 at staging) ----
    int m_tile = band * 8 + wid;
    const bool valid = (m_tile < MTILES);
    if (!valid) m_tile = MTILES - 1;           // clamp: keep barriers uniform
    const int pbase = m_tile * 16;

    const float* xsrc = x + (size_t)bc * (W448 * W448);
    const int p_idx = lane & 15;
    const int hi    = lane >> 4;
    {
        const int p  = pbase + p_idx;
        const int p0 = p / PATCH;
        const int p1 = p % PATCH;
        #pragma unroll
        for (int it = 0; it < 32; ++it) {
            const int k   = (it << 1) | hi;
            const int row = (k >> 3) * PATCH + p0;
            const int col = (k & 7)  * PATCH + p1;
            Ash[wid][p_idx * RPAD + k] = (_Float16)xsrc[row * W448 + col];
        }
    }
    __syncthreads();

    // ---- A fragments: 4x aligned ds_load_b128 (16-bit A 16x32 layout) ----
    // lane<16: h[0..7]=K c0+0..7, h[8..15]=K c0+16..23 ; lane>=16: K +8.
    const _Float16* arow = &Ash[wid][p_idx * RPAD];
    V16U a0, a1;
    a0.h[0] = *(const v8h*)(arow +  0 + hi * 8);
    a0.h[1] = *(const v8h*)(arow + 16 + hi * 8);
    a1.h[0] = *(const v8h*)(arow + 32 + hi * 8);
    a1.h[1] = *(const v8h*)(arow + 48 + hi * 8);

    float* odst = out + (size_t)bc * (W448 * W448);

    // output geometry for this lane's 8-wide runs (never crosses %56: 56%8==0)
    const int mbase = pbase + hi * 8;
    const int q0 = mbase / PATCH;
    const int q1 = mbase % PATCH;

    #pragma unroll
    for (int n0 = 0; n0 < 64; n0 += 16) {
        const int n = n0 + p_idx;
        // B fragments (32x16): lanes 0-15 hold K=c0+0..7,16..23, lanes 16-31 K+8.
        // RhT row is contiguous in k -> 2x aligned ds_load_b128 per fragment.
        const _Float16* brow = &RhT[n * RPAD];
        V16U b0, b1;
        b0.h[0] = *(const v8h*)(brow +  0 + hi * 8);
        b0.h[1] = *(const v8h*)(brow + 16 + hi * 8);
        b1.h[0] = *(const v8h*)(brow + 32 + hi * 8);
        b1.h[1] = *(const v8h*)(brow + 48 + hi * 8);

        v8f acc = {};
        acc = __builtin_amdgcn_wmma_f32_16x16x32_f16(
                  false, a0.v, false, b0.v, (short)0, acc, false, false);
        acc = __builtin_amdgcn_wmma_f32_16x16x32_f16(
                  false, a1.v, false, b1.v, (short)0, acc, false, false);

        if (valid) {
            // C/D layout: reg v -> M = v + 8*hi, N = n. The 8 values are
            // contiguous along the output column axis: 2x global_store_b128.
            const int row  = (n >> 3) * PATCH + q0;
            const int colb = (n & 7)  * PATCH + q1;
            float* dptr = odst + (size_t)row * W448 + colb;
            v4f lo = { acc[0], acc[1], acc[2], acc[3] };
            v4f hiq = { acc[4], acc[5], acc[6], acc[7] };
            *(v4f*)(dptr)     = lo;
            *(v4f*)(dptr + 4) = hiq;
        }
    }
}

// ============================ Launcher ====================================
extern "C" void kernel_launch(void* const* d_in, const int* in_sizes, int n_in,
                              void* d_out, int out_size, void* d_ws, size_t ws_size,
                              hipStream_t stream) {
    const float* x         = (const float*)d_in[0];
    const float* params    = (const float*)d_in[1];
    const int*   batch_seq = (const int*)d_in[2];
    const int*   batchsize = (const int*)d_in[3];

    float* out = (float*)d_out;
    float* Rws = (float*)d_ws;                  // 32*4096*4 = 512 KB scratch

    const int B = in_sizes[0] / (3 * W448 * W448);   // 32
    float* out_res = out + (size_t)B * 3 * W448 * W448;

    rank_chain_kernel<<<B, 256, 0, stream>>>(params, batch_seq, batchsize,
                                             Rws, out_res);
    patch_mix_kernel<<<B * 3 * BANDS, 256, 0, stream>>>(x, Rws, out);
}
